// SwitchFeedForward_8280696946828
// MI455X (gfx1250) — matrix operand order repaired and checked by reading, not compiled, not run on hardware
//
#include <hip/hip_runtime.h>

typedef __attribute__((ext_vector_type(16))) __bf16 v16bf;
typedef __attribute__((ext_vector_type(4)))  __bf16 bf4;
typedef __attribute__((ext_vector_type(8)))  float  v8f;

#define T_TOK 2048
#define Dk    1024
#define Fk    4096
#define Ek    8
#define M_BLK 128          // tokens per block tile (8 WMMA m-fragments)
#define N_BLK 128          // output cols per block (8 waves x 16)
#define K_CHUNK 64         // staged K chunk (2 WMMA k-steps)
#define XS_STRIDE 72       // padded LDS row stride (bf16 elems)

// d_out layout (floats): final[2097152] counts[8] psum[8] zero[1] pmax[2048]
#define OUT_COUNTS 2097152
#define OUT_PSUM   2097160
#define OUT_ZERO   2097168
#define OUT_PMAX   2097169

// ws layout (bytes)
#define WS_CNT     0                    // E ints (zeroed per launch)
#define WS_NTILES  64                   // 1 int
#define WS_TILE_E  128                  // 32 ints
#define WS_TILE_M  256                  // 32 ints
#define WS_PROBS   1024                 // T*E floats = 64KB
#define WS_TOKENS  (1024 + 65536)       // E*T ints  = 64KB
#define WS_H       135168               // T*F bf16 = 16MB (256B aligned)

// ---------------- routing: one wave per token ----------------
__global__ __launch_bounds__(256)
void route_kernel(const float* __restrict__ x, const float* __restrict__ sw,
                  const float* __restrict__ sb, float* __restrict__ out,
                  int* __restrict__ cnt, int* __restrict__ etok,
                  float* __restrict__ probs) {
  int wave = threadIdx.x >> 5, lane = threadIdx.x & 31;
  int t = blockIdx.x * 8 + wave;
  float acc[Ek];
#pragma unroll
  for (int e = 0; e < Ek; e++) acc[e] = 0.f;
  const float* xr = x + (size_t)t * Dk;
  for (int d = lane; d < Dk; d += 32) {
    float xv = xr[d];
    const float* wr = sw + d * Ek;
#pragma unroll
    for (int e = 0; e < Ek; e++) acc[e] += xv * wr[e];
  }
#pragma unroll
  for (int e = 0; e < Ek; e++)
#pragma unroll
    for (int off = 16; off >= 1; off >>= 1) acc[e] += __shfl_xor(acc[e], off, 32);
  if (lane == 0) {
    float l[Ek];
#pragma unroll
    for (int e = 0; e < Ek; e++) l[e] = acc[e] + sb[e];
    float m = l[0]; int am = 0;
#pragma unroll
    for (int e = 1; e < Ek; e++) if (l[e] > m) { m = l[e]; am = e; }
    float p[Ek], s = 0.f;
#pragma unroll
    for (int e = 0; e < Ek; e++) { p[e] = __expf(l[e] - m); s += p[e]; }
    float inv = 1.f / s;
#pragma unroll
    for (int e = 0; e < Ek; e++) { p[e] *= inv; probs[t * Ek + e] = p[e]; }
    out[OUT_PMAX + t] = p[am];
    int slot = atomicAdd(&cnt[am], 1);
    etok[am * T_TOK + slot] = t;
  }
}

// ---------------- finalize: psum/counts/zero + tile map ----------------
__global__ __launch_bounds__(256)
void finalize_kernel(const float* __restrict__ probs, const int* __restrict__ cnt,
                     int* __restrict__ ntile, int* __restrict__ tileE,
                     int* __restrict__ tileM, float* __restrict__ out) {
  __shared__ float red[256];
  int tid = threadIdx.x;
  for (int e = 0; e < Ek; e++) {                 // deterministic tree reduce
    float s = 0.f;
    for (int t = tid; t < T_TOK; t += 256) s += probs[t * Ek + e];
    red[tid] = s; __syncthreads();
    for (int w = 128; w >= 1; w >>= 1) {
      if (tid < w) red[tid] += red[tid + w];
      __syncthreads();
    }
    if (tid == 0) out[OUT_PSUM + e] = red[0];
    __syncthreads();
  }
  if (tid < Ek) out[OUT_COUNTS + tid] = (float)cnt[tid];
  if (tid == 0) {
    out[OUT_ZERO] = 0.0f;
    int nt = 0;
    for (int e = 0; e < Ek; e++) {
      int mt = (cnt[e] + M_BLK - 1) / M_BLK;
      for (int j = 0; j < mt; j++) { tileE[nt] = e; tileM[nt] = j; nt++; }
    }
    *ntile = nt;
  }
}

// A fragment: 16x32 bf16, lane L(half,ml): K = ks+h*16+0..15
__device__ __forceinline__ v16bf load_afrag(const __bf16* __restrict__ row, int ks, int half) {
  v16bf a;
  int base = ks + half * 16;
#pragma unroll
  for (int j = 0; j < 16; j++) a[j] = row[base + j];
  return a;
}

// B fragment: 32x16, lane(half,nl) column N=nl: K = half*16 + 0..15 (fp32->bf16)
__device__ __forceinline__ v16bf load_bfrag(const float* __restrict__ wp, int ldw) {
  v16bf b;
#pragma unroll
  for (int i = 0; i < 16; i++) b[i] = (__bf16)wp[(size_t)i * ldw];
  return b;
}

// ---------------- FFN layer 1: H = relu(X @ W1[e] + b1[e]) ----------------
__global__ __launch_bounds__(256)
void ffn1_kernel(const float* __restrict__ x, const float* __restrict__ w1,
                 const float* __restrict__ b1, const int* __restrict__ cnt,
                 const int* __restrict__ ntile, const int* __restrict__ tileE,
                 const int* __restrict__ tileM, const int* __restrict__ etok,
                 __bf16* __restrict__ H) {
  int tix = blockIdx.x;
  if (tix >= *ntile) return;
  int e = tileE[tix], mt = tileM[tix], ne = cnt[e];
  __shared__ int s_tok[M_BLK];
  __shared__ __bf16 xs[M_BLK * XS_STRIDE];
  int tid = threadIdx.x;
  if (tid < M_BLK) {
    int slot = mt * M_BLK + tid;
    if (slot >= ne) slot = ne - 1;               // clamp padding rows
    s_tok[tid] = etok[e * T_TOK + slot];
  }
  __syncthreads();

  int wave = tid >> 5, lane = tid & 31;
  int half = lane >> 4, nl = lane & 15;
  int col = blockIdx.y * N_BLK + wave * 16 + nl;
  const float* w1e = w1 + (size_t)e * Dk * Fk;

  int lm = tid >> 4, lj = (tid & 15) * 4;        // staging assignment
  int my_tok[8];
#pragma unroll
  for (int r = 0; r < 8; r++) my_tok[r] = s_tok[lm + r * 16];

  v8f acc[8];
#pragma unroll
  for (int m = 0; m < 8; m++)
#pragma unroll
    for (int i = 0; i < 8; i++) acc[m][i] = 0.f;

  for (int k0 = 0; k0 < Dk; k0 += K_CHUNK) {
#pragma unroll
    for (int r = 0; r < 8; r++) {                // stage X chunk -> bf16 LDS
      int m = lm + r * 16;
      float4 v = *(const float4*)(x + (size_t)my_tok[r] * Dk + k0 + lj);
      __bf16* dst = &xs[m * XS_STRIDE + lj];
      dst[0] = (__bf16)v.x; dst[1] = (__bf16)v.y;
      dst[2] = (__bf16)v.z; dst[3] = (__bf16)v.w;
    }
    __syncthreads();
#pragma unroll
    for (int ks = 0; ks < K_CHUNK; ks += 32) {
      const float* wp = w1e + (size_t)(k0 + ks + half * 16) * Fk + col;
      v16bf b = load_bfrag(wp, Fk);
#pragma unroll
      for (int m = 0; m < 8; m++) {
        v16bf a = load_afrag(&xs[(m * 16 + nl) * XS_STRIDE], ks, half);
        acc[m] = __builtin_amdgcn_wmma_f32_16x16x32_bf16(false, a, false, b,
                                                         (short)0, acc[m], false, false);
      }
    }
    __syncthreads();
  }
  float bias = b1[(size_t)e * Fk + col];
#pragma unroll
  for (int m = 0; m < 8; m++)
#pragma unroll
    for (int r = 0; r < 8; r++) {
      int row = m * 16 + half * 8 + r;           // C/D layout: M=r / r+8 per half-wave
      if (mt * M_BLK + row < ne) {
        float v = acc[m][r] + bias;
        H[(size_t)s_tok[row] * Fk + col] = (__bf16)(v > 0.f ? v : 0.f);
      }
    }
}

// ---------------- FFN layer 2: out = pmax * (H @ W2[e] + b2[e]) ----------------
__global__ __launch_bounds__(256)
void ffn2_kernel(const __bf16* __restrict__ H, const float* __restrict__ w2,
                 const float* __restrict__ b2, const float* __restrict__ pmax,
                 const int* __restrict__ cnt, const int* __restrict__ ntile,
                 const int* __restrict__ tileE, const int* __restrict__ tileM,
                 const int* __restrict__ etok, float* __restrict__ out) {
  int tix = blockIdx.x;
  if (tix >= *ntile) return;
  int e = tileE[tix], mt = tileM[tix], ne = cnt[e];
  __shared__ int s_tok[M_BLK];
  __shared__ __bf16 xs[M_BLK * XS_STRIDE];
  int tid = threadIdx.x;
  if (tid < M_BLK) {
    int slot = mt * M_BLK + tid;
    if (slot >= ne) slot = ne - 1;
    s_tok[tid] = etok[e * T_TOK + slot];
  }
  __syncthreads();

  int wave = tid >> 5, lane = tid & 31;
  int half = lane >> 4, nl = lane & 15;
  int col = blockIdx.y * N_BLK + wave * 16 + nl;
  const float* w2e = w2 + (size_t)e * Fk * Dk;

  int lm = tid >> 4, lj = (tid & 15) * 4;
  int my_tok[8];
#pragma unroll
  for (int r = 0; r < 8; r++) my_tok[r] = s_tok[lm + r * 16];

  v8f acc[8];
#pragma unroll
  for (int m = 0; m < 8; m++)
#pragma unroll
    for (int i = 0; i < 8; i++) acc[m][i] = 0.f;

  for (int k0 = 0; k0 < Fk; k0 += K_CHUNK) {
#pragma unroll
    for (int r = 0; r < 8; r++) {                // stage H chunk (already bf16)
      int m = lm + r * 16;
      bf4 v = *(const bf4*)(H + (size_t)my_tok[r] * Fk + k0 + lj);
      *(bf4*)&xs[m * XS_STRIDE + lj] = v;
    }
    __syncthreads();
#pragma unroll
    for (int ks = 0; ks < K_CHUNK; ks += 32) {
      const float* wp = w2e + (size_t)(k0 + ks + half * 16) * Dk + col;
      v16bf b = load_bfrag(wp, Dk);
#pragma unroll
      for (int m = 0; m < 8; m++) {
        v16bf a = load_afrag(&xs[(m * 16 + nl) * XS_STRIDE], ks, half);
        acc[m] = __builtin_amdgcn_wmma_f32_16x16x32_bf16(false, a, false, b,
                                                         (short)0, acc[m], false, false);
      }
    }
    __syncthreads();
  }
  float bias = b2[(size_t)e * Dk + col];
#pragma unroll
  for (int m = 0; m < 8; m++)
#pragma unroll
    for (int r = 0; r < 8; r++) {
      int row = m * 16 + half * 8 + r;
      if (mt * M_BLK + row < ne) {
        int tok = s_tok[row];
        out[(size_t)tok * Dk + col] = (acc[m][r] + bias) * pmax[tok];
      }
    }
}

extern "C" void kernel_launch(void* const* d_in, const int* in_sizes, int n_in,
                              void* d_out, int out_size, void* d_ws, size_t ws_size,
                              hipStream_t stream) {
  const float* x  = (const float*)d_in[0];
  const float* sw = (const float*)d_in[1];
  const float* sb = (const float*)d_in[2];
  const float* w1 = (const float*)d_in[3];
  const float* b1 = (const float*)d_in[4];
  const float* w2 = (const float*)d_in[5];
  const float* b2 = (const float*)d_in[6];
  float* out = (float*)d_out;
  char*  ws  = (char*)d_ws;

  int*    cnt   = (int*)(ws + WS_CNT);
  int*    ntile = (int*)(ws + WS_NTILES);
  int*    tileE = (int*)(ws + WS_TILE_E);
  int*    tileM = (int*)(ws + WS_TILE_M);
  float*  probs = (float*)(ws + WS_PROBS);
  int*    etok  = (int*)(ws + WS_TOKENS);
  __bf16* H     = (__bf16*)(ws + WS_H);

  hipMemsetAsync(ws, 0, 1024, stream);  // zero cnt + ntiles (graph-capturable)

  route_kernel<<<T_TOK / 8, 256, 0, stream>>>(x, sw, sb, out, cnt, etok, probs);
  finalize_kernel<<<1, 256, 0, stream>>>(probs, cnt, ntile, tileE, tileM, out);
  // max tiles = sum ceil(n_e/128) <= 16 + 7 = 23
  ffn1_kernel<<<dim3(23, Fk / N_BLK), 256, 0, stream>>>(x, w1, b1, cnt, ntile,
                                                        tileE, tileM, etok, H);
  ffn2_kernel<<<dim3(23, Dk / N_BLK), 256, 0, stream>>>(H, w2, b2, out + OUT_PMAX,
                                                        cnt, ntile, tileE, tileM,
                                                        etok, out);
}
